// TransformerBlock_45543833207026
// MI455X (gfx1250) — matrix operand order repaired and checked by reading, not compiled, not run on hardware
//
#include <hip/hip_runtime.h>
#include <hip/hip_bf16.h>
#include <math.h>

// ---------------------------------------------------------------------------
// Transformer block for MI455X (gfx1250): f16 WMMA (16x16x32) GEMMs with f32
// accumulate, TDM-staged attention panels in LDS, fp32 epilogues.
// ---------------------------------------------------------------------------

typedef _Float16 v16h __attribute__((ext_vector_type(16)));
typedef _Float16 v8h  __attribute__((ext_vector_type(8)));
typedef float    v8f  __attribute__((ext_vector_type(8)));
typedef unsigned int u32x4 __attribute__((ext_vector_type(4)));
typedef int          i32x8 __attribute__((ext_vector_type(8)));
typedef int          i32x4 __attribute__((ext_vector_type(4)));

static constexpr int Bc   = 2;
static constexpr int Pc   = 2048;
static constexpr int Cc   = 256;
static constexpr int Hc   = 8;
static constexpr int HDc  = 32;     // head dim == WMMA K. perfect fit.
static constexpr int CHc  = 1024;
static constexpr int Mr   = Bc * Pc;          // 4096 rows
static constexpr float LN_EPS = 1e-6f;

// ---------------- WMMA helpers (CDNA5 wave32 layouts) ----------------------
// A 16x32 f16: lane holds row M = lane&15; halves K = {ko..ko+7, ko+16..ko+23},
// ko = (lane>>4)*8  (per ISA 7.12.2 table).
__device__ __forceinline__ v16h load_a_tile(const _Float16* rowp) {
  int ko = ((threadIdx.x >> 4) & 1) * 8;
  v8h lo = *(const v8h*)(rowp + ko);
  v8h hi = *(const v8h*)(rowp + ko + 16);
  v16h r;
#pragma unroll
  for (int i = 0; i < 8; ++i) { r[i] = lo[i]; r[i + 8] = hi[i]; }
  return r;
}
// B 32x16 f16 (column-contiguous source): lane holds column N = lane&15,
// halves K = {ko..ko+7, ko+16..ko+23}, ko = (lane>>4)*8.
__device__ __forceinline__ v16h load_b_tile(const _Float16* colp) {
  const _Float16* p = colp + ((threadIdx.x >> 4) & 1) * 8;
  v8h lo = *(const v8h*)(p);
  v8h hi = *(const v8h*)(p + 16);
  v16h r;
#pragma unroll
  for (int i = 0; i < 8; ++i) { r[i] = lo[i]; r[i + 8] = hi[i]; }
  return r;
}
__device__ __forceinline__ v8f wmma16x16x32(v16h a, v16h b, v8f c) {
  return __builtin_amdgcn_wmma_f32_16x16x32_f16(
      /*neg_a=*/false, a, /*neg_b=*/false, b,
      /*c_mod=*/(short)0, c, /*reuse_a=*/false, /*reuse_b=*/false);
}

// ---------------- TDM: 2D tensor -> LDS (whole-panel stage) ----------------
// D# group0: [1:0]=count(1) | [63:32]=lds_addr | [120:64]=global_addr | [127:126]=type(2)
// D# group1: [17:16]=data_size(2B) | [79:48]=tensor_dim0 | [111:80]=tensor_dim1
//            | [127:112]=tile_dim0 | [143:128]=tile_dim1 | [207:160]=dim0_stride
// Toolchain probe: 6-arg builtin (uint32x4, int32x8, int32x4, int32x4, int32x8, i32 cpol)
__device__ __forceinline__ void tdm_load_2d_f16(unsigned lds_off, const void* gaddr,
                                                unsigned dim0, unsigned dim1,
                                                unsigned stride0) {
#if __has_builtin(__builtin_amdgcn_tensor_load_to_lds)
  unsigned long long ga = (unsigned long long)(uintptr_t)gaddr;
  u32x4 g0;
  g0[0] = 1u;                                           // count=1, load, user
  g0[1] = lds_off;                                      // LDS byte address
  g0[2] = (unsigned)ga;                                 // global_addr[31:0]
  g0[3] = ((unsigned)(ga >> 32) & 0x01FFFFFFu) | 0x80000000u;  // [56:32] | type=2
  i32x8 g1;
  g1[0] = (int)(1u << 16);                              // data_size = 1 -> 2 bytes
  g1[1] = (int)(dim0 << 16);                            // tensor_dim0[15:0]
  g1[2] = (int)((dim0 >> 16) | (dim1 << 16));           // td0[31:16] | td1[15:0]
  g1[3] = (int)((dim1 >> 16) | (dim0 << 16));           // td1[31:16] | tile_dim0
  g1[4] = (int)(dim1 & 0xFFFFu);                        // tile_dim1 (tile2 = 0)
  g1[5] = (int)stride0;                                 // dim0_stride[31:0]
  g1[6] = 0;
  g1[7] = 0;
  i32x4 z4 = {0, 0, 0, 0};
  i32x8 z8 = {0, 0, 0, 0, 0, 0, 0, 0};
  __builtin_amdgcn_tensor_load_to_lds(g0, g1, z4, z4, z8, 0);
#else
  (void)lds_off; (void)gaddr; (void)dim0; (void)dim1; (void)stride0;
#endif
}

// ---------------- weight convert + transpose (f32 KxN -> f16 NxK) ----------
__global__ void tcvt_kernel(const float* __restrict__ W, _Float16* __restrict__ Wt,
                            int K, int N) {
  size_t i = (size_t)blockIdx.x * blockDim.x + threadIdx.x;
  if (i >= (size_t)K * N) return;
  int k = (int)(i / N), n = (int)(i % N);
  Wt[(size_t)n * K + k] = (_Float16)W[i];
}

// ---------------- LayerNorm: one wave per row ------------------------------
__global__ void ln_kernel(const float* __restrict__ X, const float* __restrict__ g,
                          const float* __restrict__ b, _Float16* __restrict__ out,
                          int Ccols) {
  int row = blockIdx.x, lane = threadIdx.x;
  const float* xr = X + (size_t)row * Ccols;
  float s = 0.f, ss = 0.f;
  for (int j = lane; j < Ccols; j += 32) { float v = xr[j]; s += v; ss += v * v; }
#pragma unroll
  for (int off = 16; off > 0; off >>= 1) {
    s  += __shfl_xor(s,  off, 32);
    ss += __shfl_xor(ss, off, 32);
  }
  float mu = s / Ccols;
  float var = ss / Ccols - mu * mu;
  float rs = rsqrtf(var + LN_EPS);
  for (int j = lane; j < Ccols; j += 32)
    out[(size_t)row * Ccols + j] = (_Float16)((xr[j] - mu) * rs * g[j] + b[j]);
}

// ---------------- generic WMMA GEMM, one wave per 16x64 tile ---------------
// A: f16 row-major MxK.  Bt: f16 "column major" = NxK row-major.
// mode 0: Q scatter, 1: K scatter, 2: V scatter (+fuse_v), 3: proj+resid f32,
// mode 4/5: plain f32 store.
__global__ __launch_bounds__(32)
void gemm_kernel(const _Float16* __restrict__ A, const _Float16* __restrict__ Bt,
                 const float* __restrict__ bias, int N, int K, int mode,
                 const float* __restrict__ resid, float* __restrict__ out32,
                 _Float16* __restrict__ out16, float* __restrict__ fv_out) {
  int tm = blockIdx.x, tn = blockIdx.y;                 // tn covers 64 columns
  int lane = threadIdx.x, col = lane & 15, half = lane >> 4;

  const _Float16* arow  = A  + (size_t)(tm * 16 + col) * K;
  const _Float16* bcol0 = Bt + (size_t)(tn * 64 + col) * K;
  v8f acc[4] = {};
  for (int k0 = 0; k0 < K; k0 += 32) {
    if (k0 + 32 < K) {
      __builtin_prefetch(arow + k0 + 32, 0, 1);         // global_prefetch_b8
      __builtin_prefetch(bcol0 + k0 + 32, 0, 1);
    }
    v16h a = load_a_tile(arow + k0);                    // A reused by 4 WMMAs
#pragma unroll
    for (int nt = 0; nt < 4; ++nt) {
      v16h b = load_b_tile(bcol0 + (size_t)nt * 16 * K + k0);
      acc[nt] = wmma16x16x32(a, b, acc[nt]);
    }
  }

#pragma unroll
  for (int nt = 0; nt < 4; ++nt) {
    int n = tn * 64 + nt * 16 + col;
    float bv = bias ? bias[n] : 0.f;
#pragma unroll
    for (int r = 0; r < 8; ++r) {
      int m = tm * 16 + r + 8 * half;                   // D layout: lane=col, VGPR r=row
      float val = acc[nt][r] + bv;
      size_t idx = (size_t)m * N + n;
      if (mode <= 2) {                                  // q/k/v: c = d*H + h
        int bb = m / Pc, p = m % Pc;
        int hh = n & (Hc - 1), d = n >> 3;
        if (mode == 0 || mode == 1) {
          out16[(((size_t)(bb * Hc + hh)) * Pc + p) * HDc + d] = (_Float16)val;
        } else {
          out16[(((size_t)(bb * Hc + hh)) * HDc + d) * Pc + p] = (_Float16)val; // V^T
          if (hh == Hc - 1) fv_out[((size_t)bb * Pc + p) * HDc + d] = val;
        }
      } else if (mode == 3) {
        out32[idx] = val + resid[idx];
      } else {
        out32[idx] = val;
      }
    }
  }
}

// ---------------- attention ------------------------------------------------
// Grid: 64 blocks = (b,h) x 4 q-quarters; block = 4 waves, 8 q-tiles per wave.
// K panel (2048x32 f16, 128KB) and V^T panel (32x2048 f16, 128KB) are staged
// once per block into LDS by the Tensor Data Mover, then both softmax passes
// and the p@V WMMAs read LDS.
__global__ __launch_bounds__(128)
void attn_kernel(const _Float16* __restrict__ qb, const _Float16* __restrict__ kb,
                 const _Float16* __restrict__ vt, _Float16* __restrict__ attnin,
                 float* __restrict__ fw_out) {
  extern __shared__ __align__(16) char dynsmem[];
  _Float16* kpan  = (_Float16*)dynsmem;                    // (P, HD) row-major
  _Float16* vpan  = kpan + (size_t)Pc * HDc;               // (HD, P) row-major
  _Float16* pbufs = vpan + (size_t)Pc * HDc;               // 4 waves x 16x32

  int blk  = blockIdx.x;
  int qgrp = blk & 3;            // quarter of the 128 q tiles
  int bh   = blk >> 2;           // 0..15
  int hd   = bh & (Hc - 1);
  int b    = bh >> 3;

  const _Float16* qg = qb + (size_t)bh * Pc * HDc;
  const _Float16* kg = kb + (size_t)bh * Pc * HDc;
  const _Float16* vg = vt + (size_t)bh * Pc * HDc;

  int tid = threadIdx.x, wave = tid >> 5, lane = tid & 31;
  int col = lane & 15, half = lane >> 4;
  const float scale = 0.1767766952966369f;  // 1/sqrt(32)

  // ---- stage K and V^T panels via TDM (wave 0 issues, all waves barrier) --
#if __has_builtin(__builtin_amdgcn_tensor_load_to_lds)
  if (wave == 0) {
    tdm_load_2d_f16((unsigned)(uintptr_t)kpan, kg, HDc, Pc, HDc);  // dim0=32, dim1=2048
    tdm_load_2d_f16((unsigned)(uintptr_t)vpan, vg, Pc, HDc, Pc);   // dim0=2048, dim1=32
    __builtin_amdgcn_s_wait_tensorcnt(0);
  }
#else
  for (size_t i = tid; i < (size_t)Pc * HDc; i += blockDim.x) {
    kpan[i] = kg[i];
    vpan[i] = vg[i];
  }
#endif
  __syncthreads();

  _Float16* pbuf = pbufs + wave * (16 * 32);

  for (int qi = 0; qi < 8; ++qi) {
    int qt = qgrp * 32 + wave * 8 + qi;

    // q tile lives in registers for both passes (K = HD = 32, one WMMA deep)
    v16h aq = load_a_tile(qg + (size_t)(qt * 16 + col) * HDc);

    // ---- pass 1: online row max / sum-exp over all 128 key tiles ----
    float m8[8], s8[8];
#pragma unroll
    for (int r = 0; r < 8; ++r) { m8[r] = -3.0e38f; s8[r] = 0.f; }

    for (int kt = 0; kt < Pc / 16; ++kt) {
      v16h bk = load_b_tile(kpan + (size_t)(kt * 16 + col) * HDc);   // ds_load
      v8f d = {};
      d = wmma16x16x32(aq, bk, d);
#pragma unroll
      for (int r = 0; r < 8; ++r) {
        float x = d[r] * scale;
        float mx = x;
#pragma unroll
        for (int off = 1; off < 16; off <<= 1) mx = fmaxf(mx, __shfl_xor(mx, off, 32));
        float mn = fmaxf(m8[r], mx);
        float e = __expf(x - mn);
#pragma unroll
        for (int off = 1; off < 16; off <<= 1) e += __shfl_xor(e, off, 32);
        s8[r] = s8[r] * __expf(m8[r] - mn) + e;
        m8[r] = mn;
      }
    }
    float inv_s[8];
#pragma unroll
    for (int r = 0; r < 8; ++r) inv_s[r] = 1.f / s8[r];

    // ---- pass 2: recompute, normalize, emit head-7 weights, p @ V ----
    v8f acc0 = {}, acc1 = {};
    for (int kt = 0; kt < Pc / 16; ++kt) {
      v16h bk = load_b_tile(kpan + (size_t)(kt * 16 + col) * HDc);
      v8f d = {};
      d = wmma16x16x32(aq, bk, d);
      int slot = kt & 1;
#pragma unroll
      for (int r = 0; r < 8; ++r) {
        int row = r + 8 * half;
        float p = __expf(d[r] * scale - m8[r]) * inv_s[r];
        if (hd == Hc - 1)
          fw_out[((size_t)b * Pc + qt * 16 + row) * Pc + kt * 16 + col] = p;
        pbuf[row * 32 + slot * 16 + col] = (_Float16)p;
      }
      if (slot == 1) {
        // per-wave pbuf region: DS ops are in-order within the wave; the
        // compiler inserts the dscnt wait for the store->load dependency.
        int kbase = (kt - 1) * 16;
        v16h ap = load_a_tile(pbuf + (size_t)col * 32);                 // 16x32 p
        v16h b0 = load_b_tile(vpan + (size_t)col * Pc + kbase);         // d = 0..15
        v16h b1 = load_b_tile(vpan + (size_t)(16 + col) * Pc + kbase);  // d = 16..31
        acc0 = wmma16x16x32(ap, b0, acc0);
        acc1 = wmma16x16x32(ap, b1, acc1);
      }
    }

    // head.transpose(0,2,1,3).reshape: c = h*HD + d
#pragma unroll
    for (int r = 0; r < 8; ++r) {
      int m = qt * 16 + r + 8 * half;
      size_t base = ((size_t)b * Pc + m) * Cc + hd * HDc;
      attnin[base + col]      = (_Float16)acc0[r];
      attnin[base + 16 + col] = (_Float16)acc1[r];
    }
  }
}

// ---------------- BatchNorm reduce: thread = channel (coalesced) -----------
__global__ void bn_reduce_kernel(const float* __restrict__ X, int rows, int ch,
                                 float* __restrict__ mu, float* __restrict__ var) {
  int c = blockIdx.x * blockDim.x + threadIdx.x;
  if (c >= ch) return;
  float s = 0.f, ss = 0.f;
  for (int r = 0; r < rows; ++r) {
    float v = X[(size_t)r * ch + c];
    s += v; ss += v * v;
  }
  float m = s / rows;
  mu[c] = m;
  var[c] = ss / rows - m * m;
}

// ---------------- BN + exact GELU -> f16 activations -----------------------
__global__ void bn_gelu_kernel(const float* __restrict__ X, const float* __restrict__ mu,
                               const float* __restrict__ var, const float* __restrict__ g,
                               const float* __restrict__ b, _Float16* __restrict__ out,
                               int rows, int ch) {
  size_t i = (size_t)blockIdx.x * blockDim.x + threadIdx.x;
  if (i >= (size_t)rows * ch) return;
  int c = (int)(i % ch);
  float v = (X[i] - mu[c]) * rsqrtf(var[c] + LN_EPS) * g[c] + b[c];
  float ge = 0.5f * v * (1.f + erff(v * 0.70710678118654752f));
  out[i] = (_Float16)ge;
}

// ---------------- final: out = x1 + BN2(y) ---------------------------------
__global__ void bn_add_kernel(const float* __restrict__ Y, const float* __restrict__ mu,
                              const float* __restrict__ var, const float* __restrict__ g,
                              const float* __restrict__ b, const float* __restrict__ x1,
                              float* __restrict__ out, int rows, int ch) {
  size_t i = (size_t)blockIdx.x * blockDim.x + threadIdx.x;
  if (i >= (size_t)rows * ch) return;
  int c = (int)(i % ch);
  float v = (Y[i] - mu[c]) * rsqrtf(var[c] + LN_EPS) * g[c] + b[c];
  out[i] = x1[i] + v;
}

// ---------------------------------------------------------------------------
extern "C" void kernel_launch(void* const* d_in, const int* in_sizes, int n_in,
                              void* d_out, int out_size, void* d_ws, size_t ws_size,
                              hipStream_t stream) {
  (void)in_sizes; (void)n_in; (void)out_size; (void)ws_size;

  const float* x     = (const float*)d_in[0];
  const float* ln1_g = (const float*)d_in[1];
  const float* ln1_b = (const float*)d_in[2];
  const float* ln2_g = (const float*)d_in[3];
  const float* ln2_b = (const float*)d_in[4];
  const float* Wq = (const float*)d_in[5];  const float* bq = (const float*)d_in[6];
  const float* Wk = (const float*)d_in[7];  const float* bk = (const float*)d_in[8];
  const float* Wv = (const float*)d_in[9];  const float* bv = (const float*)d_in[10];
  const float* Wp = (const float*)d_in[11]; const float* bp = (const float*)d_in[12];
  const float* W1 = (const float*)d_in[13]; const float* b1 = (const float*)d_in[14];
  const float* bn1_g = (const float*)d_in[15]; const float* bn1_b = (const float*)d_in[16];
  const float* W2 = (const float*)d_in[17]; const float* b2 = (const float*)d_in[18];
  const float* bn2_g = (const float*)d_in[19]; const float* bn2_b = (const float*)d_in[20];

  float* out_x  = (float*)d_out;                               // (B,P,C)
  float* out_fw = out_x + (size_t)Bc * Pc * Cc;                // (B,P,P)
  float* out_fv = out_fw + (size_t)Bc * Pc * Pc;               // (B,P,HD)

  // workspace carve
  char* wp = (char*)d_ws;
  auto carve = [&](size_t bytes) -> void* {
    void* r = (void*)wp;
    wp += (bytes + 255) & ~(size_t)255;
    return r;
  };
  _Float16* lnx    = (_Float16*)carve((size_t)Mr * Cc * 2);
  _Float16* WqT    = (_Float16*)carve((size_t)Cc * Cc * 2);
  _Float16* WkT    = (_Float16*)carve((size_t)Cc * Cc * 2);
  _Float16* WvT    = (_Float16*)carve((size_t)Cc * Cc * 2);
  _Float16* WpT    = (_Float16*)carve((size_t)Cc * Cc * 2);
  _Float16* W1T    = (_Float16*)carve((size_t)Cc * CHc * 2);
  _Float16* W2T    = (_Float16*)carve((size_t)CHc * Cc * 2);
  _Float16* qbuf   = (_Float16*)carve((size_t)Bc * Hc * Pc * HDc * 2);
  _Float16* kbuf   = (_Float16*)carve((size_t)Bc * Hc * Pc * HDc * 2);
  _Float16* vtbuf  = (_Float16*)carve((size_t)Bc * Hc * HDc * Pc * 2);
  _Float16* attnin = (_Float16*)carve((size_t)Mr * Cc * 2);
  float*    x1     = (float*)   carve((size_t)Mr * Cc * 4);
  _Float16* ln2x   = (_Float16*)carve((size_t)Mr * Cc * 2);
  float*    hpre   = (float*)   carve((size_t)Mr * CHc * 4);
  _Float16* hbuf   = (_Float16*)carve((size_t)Mr * CHc * 2);
  float*    ybuf   = (float*)   carve((size_t)Mr * Cc * 4);
  float*    mu1    = (float*)   carve((size_t)CHc * 4);
  float*    var1   = (float*)   carve((size_t)CHc * 4);
  float*    mu2    = (float*)   carve((size_t)Cc * 4);
  float*    var2   = (float*)   carve((size_t)Cc * 4);

  // 1) weights -> f16 transposed (NxK)
  auto nblk = [](size_t n) { return (unsigned)((n + 255) / 256); };
  tcvt_kernel<<<nblk((size_t)Cc * Cc), 256, 0, stream>>>(Wq, WqT, Cc, Cc);
  tcvt_kernel<<<nblk((size_t)Cc * Cc), 256, 0, stream>>>(Wk, WkT, Cc, Cc);
  tcvt_kernel<<<nblk((size_t)Cc * Cc), 256, 0, stream>>>(Wv, WvT, Cc, Cc);
  tcvt_kernel<<<nblk((size_t)Cc * Cc), 256, 0, stream>>>(Wp, WpT, Cc, Cc);
  tcvt_kernel<<<nblk((size_t)Cc * CHc), 256, 0, stream>>>(W1, W1T, Cc, CHc);
  tcvt_kernel<<<nblk((size_t)CHc * Cc), 256, 0, stream>>>(W2, W2T, CHc, Cc);

  // 2) LN1
  ln_kernel<<<Mr, 32, 0, stream>>>(x, ln1_g, ln1_b, lnx, Cc);

  // 3) QKV GEMMs (wave = 16x64 tile)
  dim3 gq(Mr / 16, Cc / 64);
  gemm_kernel<<<gq, 32, 0, stream>>>(lnx, WqT, bq, Cc, Cc, 0, nullptr, nullptr, qbuf, nullptr);
  gemm_kernel<<<gq, 32, 0, stream>>>(lnx, WkT, bk, Cc, Cc, 1, nullptr, nullptr, kbuf, nullptr);
  gemm_kernel<<<gq, 32, 0, stream>>>(lnx, WvT, bv, Cc, Cc, 2, nullptr, nullptr, vtbuf, out_fv);

  // 4) attention: TDM-staged K/V^T panels in LDS, two-pass softmax, p@V WMMA
  size_t attn_lds = (size_t)Pc * HDc * 2 * 2 + 4 * 16 * 32 * 2;   // 256KB + 4KB
  attn_kernel<<<16 * 4, 128, attn_lds, stream>>>(qbuf, kbuf, vtbuf, attnin, out_fw);

  // 5) projection + residual -> x1
  gemm_kernel<<<gq, 32, 0, stream>>>(attnin, WpT, bp, Cc, Cc, 3, x, x1, nullptr, nullptr);

  // 6) LN2
  ln_kernel<<<Mr, 32, 0, stream>>>(x1, ln2_g, ln2_b, ln2x, Cc);

  // 7) MLP1 -> hpre (f32, pre-BN)
  dim3 g1(Mr / 16, CHc / 64);
  gemm_kernel<<<g1, 32, 0, stream>>>(ln2x, W1T, b1, CHc, Cc, 4, nullptr, hpre, nullptr, nullptr);

  // 8) BN1 stats + BN1/GELU -> hbuf (f16)
  bn_reduce_kernel<<<(CHc + 255) / 256, 256, 0, stream>>>(hpre, Mr, CHc, mu1, var1);
  bn_gelu_kernel<<<nblk((size_t)Mr * CHc), 256, 0, stream>>>(hpre, mu1, var1, bn1_g, bn1_b,
                                                             hbuf, Mr, CHc);

  // 9) MLP2 -> ybuf (f32, pre-BN)
  dim3 g2(Mr / 16, Cc / 64);
  gemm_kernel<<<g2, 32, 0, stream>>>(hbuf, W2T, b2, Cc, CHc, 5, nullptr, ybuf, nullptr, nullptr);

  // 10) BN2 stats + residual fuse -> final x output
  bn_reduce_kernel<<<(Cc + 255) / 256, 256, 0, stream>>>(ybuf, Mr, Cc, mu2, var2);
  bn_add_kernel<<<nblk((size_t)Mr * Cc), 256, 0, stream>>>(ybuf, mu2, var2, bn2_g, bn2_b,
                                                           x1, out_x, Mr, Cc);
}
